// DiagonalLSTM_76922864271419
// MI455X (gfx1250) — matrix operand order repaired and checked by reading, not compiled, not run on hardware
//
#include <hip/hip_runtime.h>
#include <hip/hip_bf16.h>
#include <stdint.h>

// ---------------------------------------------------------------------------
// CDNA5 (gfx1250) diagonal-LSTM.  wave32, WMMA f16 -> f32 accumulate,
// async global->LDS double-buffered input staging.
// ---------------------------------------------------------------------------

typedef __attribute__((ext_vector_type(16))) _Float16 v16h;
typedef __attribute__((ext_vector_type(8)))  float    v8f;
typedef __attribute__((ext_vector_type(4)))  int      v4i;

#define AS1 __attribute__((address_space(1)))
#define AS3 __attribute__((address_space(3)))

#if defined(__has_builtin)
#  if __has_builtin(__builtin_amdgcn_global_load_async_to_lds_b128)
#    define HAVE_ASYNC_LDS 1
#  endif
#endif
#ifndef HAVE_ASYNC_LDS
#  define HAVE_ASYNC_LDS 0
#endif

#if HAVE_ASYNC_LDS
#  if defined(__has_builtin) && __has_builtin(__builtin_amdgcn_s_wait_asynccnt)
#    define WAIT_ASYNC(n) __builtin_amdgcn_s_wait_asynccnt(n)
#  else
#    define WAIT_ASYNC(n) asm volatile("s_wait_asynccnt " #n ::: "memory")
#  endif
#else
#  define WAIT_ASYNC(n) do { } while (0)
#endif

// D = A(16x32 f16) * B(32x16 f16) + C(16x16 f32)
__device__ __forceinline__ v8f wmma_f16(v16h a, v16h b, v8f c) {
  return __builtin_amdgcn_wmma_f32_16x16x32_f16(
      /*neg_a=*/false, a, /*neg_b=*/false, b,
      /*c_mod=*/(short)0, c, /*reuse_a=*/false, /*reuse_b=*/false);
}

// A-matrix 16x32 f16 fragment from row-major LDS (ld = row stride in halfs).
// ISA layout: lane L -> M = L&15 ; VGPR v, slot s -> K = (v>=4?16:0)+(v&3)*2+s+(L>>4)*8
__device__ __forceinline__ v16h load_a_frag(const _Float16* base, int ld, int lane) {
  const _Float16* row = base + (size_t)(lane & 15) * ld;
  const int kh = (lane >> 4) << 3;
  v16h a;
#pragma unroll
  for (int v = 0; v < 8; ++v) {
    int k = ((v & 4) << 2) + ((v & 3) << 1) + kh;
    a[2 * v + 0] = row[k + 0];
    a[2 * v + 1] = row[k + 1];
  }
  return a;
}

// B fragments pre-swizzled in ws: lane L of tile T holds 16 contiguous halfs at
// fB[(T*32+L)*16 + e]; element e is B[kt*32+(e>>3)*16+(L>>4)*8+(e&7)][nt*16+(L&15)].
__device__ __forceinline__ v16h load_b_frag(const _Float16* fB, int tile, int lane) {
  return *(const v16h*)(fB + (((size_t)tile * 32 + lane) << 4));
}

// ---------------------------------------------------------------------------
// Kernel 1: transpose input (B,C,H,W) f32 -> xT[b][p][w][c] f16 (LDS-tiled)
// ---------------------------------------------------------------------------
__global__ void __launch_bounds__(256) prep_x_kernel(const float* __restrict__ in,
                                                     _Float16* __restrict__ xT) {
  __shared__ float tile[64][65];
  const int bp = blockIdx.x;            // b*64 + p
  const int b = bp >> 6, p = bp & 63;
  const float* src = in + (size_t)b * 262144 + (size_t)p * 64;  // + c*4096 + w
#pragma unroll
  for (int k = 0; k < 16; ++k) {
    int idx = k * 256 + threadIdx.x;
    int c = idx >> 6, w = idx & 63;
    tile[c][w] = src[(size_t)c * 4096 + w];
  }
  __syncthreads();
  _Float16* dst = xT + (size_t)bp * 4096;  // [w][c]
#pragma unroll
  for (int k = 0; k < 16; ++k) {
    int idx = k * 256 + threadIdx.x;
    int w = idx >> 6, c = idx & 63;
    dst[(size_t)w * 64 + c] = (_Float16)tile[c][w];
  }
}

// ---------------------------------------------------------------------------
// Kernel 2: pre-swizzle all weight matrices into WMMA B-fragment order (f16)
//   fS: [Ws0|Ws1]^T  K=128 N=256 (64 tiles)   fC: [Wc0|Wc1]^T K=128 N=64 (16)
//   fI: w_i2s^T      K=64  N=256 (32 tiles)   fU: w_up^T     K=64  N=128 (16)
// ---------------------------------------------------------------------------
__global__ void __launch_bounds__(256) prep_w_kernel(
    const float* __restrict__ w_i2s, const float* __restrict__ w_s2s,
    const float* __restrict__ w_c2c, const float* __restrict__ w_up,
    _Float16* __restrict__ fS, _Float16* __restrict__ fC,
    _Float16* __restrict__ fI, _Float16* __restrict__ fU) {
  int gid = blockIdx.x * 256 + threadIdx.x;  // 65536 total fragment elements
  _Float16* dst;
  int NT, which, idx = gid;
  if (idx < 32768)        { dst = fS; NT = 16; which = 0; }
  else if (idx < 40960)   { idx -= 32768; dst = fC; NT = 4;  which = 1; }
  else if (idx < 57344)   { idx -= 40960; dst = fI; NT = 16; which = 2; }
  else                    { idx -= 57344; dst = fU; NT = 8;  which = 3; }
  const int e    = idx & 15;
  const int lane = (idx >> 4) & 31;
  const int tile = idx >> 9;
  const int kt = tile / NT, nt = tile % NT;
  const int k = kt * 32 + ((e >> 3) << 4) + ((lane >> 4) << 3) + (e & 7);   // 2*v+s == e
  const int o = nt * 16 + (lane & 15);
  float val;
  if (which == 0)      val = (k < 64) ? w_s2s[o * 128 + k * 2]
                                      : w_s2s[o * 128 + (k - 64) * 2 + 1];
  else if (which == 1) val = (k < 64) ? w_c2c[o * 128 + k * 2]
                                      : w_c2c[o * 128 + (k - 64) * 2 + 1];
  else if (which == 2) val = w_i2s[o * 64 + k];
  else                 val = w_up[o * 64 + k];
  dst[idx] = (_Float16)val;
}

// ---------------------------------------------------------------------------
// Kernel 3: persistent diagonal-LSTM recurrence. 1 workgroup per batch (32),
// 256 threads = 8 waves. 127 serial steps, all state in LDS, all weight
// B-fragments resident in VGPRs, async-prefetched x tiles, 2 barriers/step.
// ---------------------------------------------------------------------------
__global__ void __launch_bounds__(256) lstm_main_kernel(
    const _Float16* __restrict__ xT, const _Float16* __restrict__ fS,
    const _Float16* __restrict__ fC, const _Float16* __restrict__ fI,
    const float* __restrict__ b_i2s, const float* __restrict__ b_s2s,
    const float* __restrict__ b_c2c, _Float16* __restrict__ hid) {
  // LDS state (~130 KB, well under 320 KB/WGP)
  __shared__ __align__(16) _Float16 hcat[64][128];  // [p][0:64]=h_shift  [p][64:128]=h_prev
  __shared__ __align__(16) _Float16 ccat[64][128];  // same for cell state
  __shared__ __align__(16) _Float16 xA[2][64][64];  // double-buffered i2s A-tile
  __shared__ __align__(16) uint4 xdump;             // async sink for out-of-band rows
  __shared__ float S[64][256];                      // sigmoid(i2s + s2s + biases)
  __shared__ float CC[64][64];                      // c2c + bias
  __shared__ float bias_si[256];
  __shared__ float bias_c[64];

  const int b    = blockIdx.x;
  const int wv   = threadIdx.x >> 5;   // 0..7
  const int lane = threadIdx.x & 31;
  const int lm   = lane & 15;
  const int lhi8 = (lane >> 4) << 3;

  const _Float16* xbase = xT + (size_t)b * 262144;  // [p][w][c]

  // ---- init: zero state, fold biases -------------------------------------
  for (int idx = threadIdx.x; idx < 64 * 128; idx += 256) {
    ((_Float16*)hcat)[idx] = (_Float16)0.f;
    ((_Float16*)ccat)[idx] = (_Float16)0.f;
  }
  if (threadIdx.x < 256) bias_si[threadIdx.x] = b_s2s[threadIdx.x] + b_i2s[threadIdx.x];
  if (threadIdx.x < 64)  bias_c[threadIdx.x]  = b_c2c[threadIdx.x];

  // ---- load resident B fragments (stay in VGPRs for all 127 steps) -------
  v16h bS8[8], bI4[4], bC8[8];
#pragma unroll
  for (int i = 0; i < 2; ++i) {
    const int nt  = 2 * wv + i;
    const int ntc = (2 * wv + i) & 3;
#pragma unroll
    for (int kt = 0; kt < 4; ++kt) bS8[i * 4 + kt] = load_b_frag(fS, kt * 16 + nt, lane);
#pragma unroll
    for (int kt = 0; kt < 2; ++kt) bI4[i * 2 + kt] = load_b_frag(fI, kt * 16 + nt, lane);
#pragma unroll
    for (int kt = 0; kt < 4; ++kt) bC8[i * 4 + kt] = load_b_frag(fC, kt * 4 + ntc, lane);
  }

  // ---- staging helper: bring step-t x tile into xA[t&1] -------------------
  // Async path: every chunk issues exactly one async op (uniform ASYNCcnt of 2
  // per wave per stage); out-of-band chunks are routed to xdump and the real
  // row is zeroed with an ordinary LDS store.
  auto stage_x = [&](int t) {
    for (int idx = threadIdx.x; idx < 512; idx += 256) {
      const int p = idx >> 3, ch = idx & 7;
      const int w = t - p;
      const bool valid = (0 <= w && w < 64);
#if HAVE_ASYNC_LDS
      const _Float16* src = xbase + ((size_t)p * 64 + (valid ? w : 0)) * 64 + ch * 8;
      v4i* dstp = valid ? (v4i*)&xA[t & 1][p][ch * 8] : (v4i*)&xdump;
      __builtin_amdgcn_global_load_async_to_lds_b128(
          (AS1 v4i*)(v4i*)src, (AS3 v4i*)dstp, 0, 0);
      if (!valid) *(uint4*)&xA[t & 1][p][ch * 8] = make_uint4(0u, 0u, 0u, 0u);
#else
      uint4 v = make_uint4(0u, 0u, 0u, 0u);
      if (valid) v = *(const uint4*)(xbase + ((size_t)p * 64 + w) * 64 + ch * 8);
      *(uint4*)&xA[t & 1][p][ch * 8] = v;
#endif
    }
  };

  stage_x(0);  // prologue prefetch for t = 0

  for (int t = 0; t < 127; ++t) {
    // -- prefetch next tile, then wait for the current one -----------------
    if (t + 1 < 127) {
      stage_x(t + 1);
      WAIT_ASYNC(2);   // 2 newest (t+1) may remain; step-t loads are done
    } else {
      WAIT_ASYNC(0);
    }
    __syncthreads();   // x tile + previous step's state updates visible

    const _Float16(*xcur)[64] = xA[t & 1];

    // -- WMMA GEMMs ---------------------------------------------------------
    // gate pre-activations: S = sigmoid( x@Wi2s^T + [hs|hp]@[Ws0|Ws1]^T + b )
#pragma unroll
    for (int mt = 0; mt < 4; ++mt) {
      v16h aX[2], aH[4];
#pragma unroll
      for (int kt = 0; kt < 2; ++kt) aX[kt] = load_a_frag(&xcur[mt * 16][kt * 32], 64, lane);
#pragma unroll
      for (int kt = 0; kt < 4; ++kt) aH[kt] = load_a_frag(&hcat[mt * 16][kt * 32], 128, lane);
#pragma unroll
      for (int i = 0; i < 2; ++i) {
        const int n0 = (2 * wv + i) * 16;
        v8f acc = {};
#pragma unroll
        for (int kt = 0; kt < 2; ++kt) acc = wmma_f16(aX[kt], bI4[i * 2 + kt], acc);
#pragma unroll
        for (int kt = 0; kt < 4; ++kt) acc = wmma_f16(aH[kt], bS8[i * 4 + kt], acc);
        const float bias = bias_si[n0 + lm];
#pragma unroll
        for (int r = 0; r < 8; ++r) {
          float v = acc[r] + bias;
          S[mt * 16 + lhi8 + r][n0 + lm] = 1.f / (1.f + __expf(-v));
        }
      }
    }
    // cell path: CC = [cs|cp]@[Wc0|Wc1]^T + b_c2c
#pragma unroll
    for (int i = 0; i < 2; ++i) {
      const int tix = 2 * wv + i;
      const int mtc = tix >> 2, ntc = tix & 3;
      v8f acc = {};
#pragma unroll
      for (int kt = 0; kt < 4; ++kt) {
        v16h a = load_a_frag(&ccat[mtc * 16][kt * 32], 128, lane);
        acc = wmma_f16(a, bC8[i * 4 + kt], acc);
      }
      const float bias = bias_c[ntc * 16 + lm];
#pragma unroll
      for (int r = 0; r < 8; ++r)
        CC[mtc * 16 + lhi8 + r][ntc * 16 + lm] = acc[r] + bias;
    }
    __syncthreads();

    // -- gates (reference's flattened-axis split), state update, banded
    //    hidden-state emission ---------------------------------------------
#pragma unroll
    for (int i = 0; i < 16; ++i) {
      const int e = i * 256 + threadIdx.x;  // 0..4095
      const int p = e >> 6, j = e & 63;
      const int q = p >> 2, o = ((p & 3) << 6) + j;
      const float ig = S[q][o];
      const float gg = S[16 + q][o];
      const float fg = S[32 + q][o];
      const float og = S[48 + q][o];
      const float c  = fg * CC[p][j] + ig * gg;
      const float hh = og * tanhf(c);
      const _Float16 chf = (_Float16)c;
      const _Float16 hhf = (_Float16)hh;
      ccat[p][64 + j] = chf;                  // c_prev for next step
      hcat[p][64 + j] = hhf;                  // h_prev for next step
      if (p < 63) { ccat[p + 1][j] = chf; hcat[p + 1][j] = hhf; }  // shifted
      const int w = t - p;
      if (0 <= w && w < 64)
        hid[(((size_t)b * 64 + p) * 64 + w) * 64 + j] = hhf;       // unskewed
    }
    // no trailing barrier: next iteration's top-of-loop barrier orders these
    // writes before the next step's reads; the staging buffer written next
    // iteration was last read before the post-GEMM barrier of this step.
  }
}

// ---------------------------------------------------------------------------
// Kernel 4: up-projection out[b][oc][p][w] = hid[b][p][w][:] @ w_up^T + b_up
// One block per 16-row M-tile; 8 waves cover N=128.
// ---------------------------------------------------------------------------
__global__ void __launch_bounds__(256) up_proj_kernel(
    const _Float16* __restrict__ hid, const _Float16* __restrict__ fU,
    const float* __restrict__ b_up, float* __restrict__ out) {
  __shared__ __align__(16) _Float16 tile[16][64];
  const int m0 = blockIdx.x * 16;               // global row = ((b*64+p)*64+w)
  const int b = m0 >> 12, rem = m0 & 4095;
  const int p = rem >> 6, w0 = rem & 63;

  {  // coalesced 16x64 f16 tile load (each thread: 4 halfs)
    const int idx = threadIdx.x * 4;
    const int rr = idx >> 6, cc = idx & 63;
    *(uint2*)&tile[rr][cc] = *(const uint2*)(hid + ((size_t)m0 + rr) * 64 + cc);
  }
  __syncthreads();

  const int wv = threadIdx.x >> 5, lane = threadIdx.x & 31;
  const int lm = lane & 15, lhi8 = (lane >> 4) << 3;
  v8f acc = {};
#pragma unroll
  for (int kt = 0; kt < 2; ++kt) {
    v16h a  = load_a_frag(&tile[0][kt * 32], 64, lane);
    v16h bf = load_b_frag(fU, kt * 8 + wv, lane);
    acc = wmma_f16(a, bf, acc);
  }
  const int oc = wv * 16 + lm;
  const float bias = b_up[oc];
  float* dst = out + (((size_t)b * 128 + oc) * 64 + p) * 64 + (w0 + lhi8);
#pragma unroll
  for (int r = 0; r < 8; ++r) dst[r] = acc[r] + bias;
}

// ---------------------------------------------------------------------------
extern "C" void kernel_launch(void* const* d_in, const int* in_sizes, int n_in,
                              void* d_out, int out_size, void* d_ws, size_t ws_size,
                              hipStream_t stream) {
  const float* inputs = (const float*)d_in[0];
  const float* w_i2s  = (const float*)d_in[1];
  const float* b_i2s  = (const float*)d_in[2];
  const float* w_s2s  = (const float*)d_in[3];
  const float* b_s2s  = (const float*)d_in[4];
  const float* w_c2c  = (const float*)d_in[5];
  const float* b_c2c  = (const float*)d_in[6];
  const float* w_up   = (const float*)d_in[7];
  const float* b_up   = (const float*)d_in[8];

  char* ws = (char*)d_ws;
  _Float16* xT  = (_Float16*)(ws);                          // 16 MB
  _Float16* hid = (_Float16*)(ws + ((size_t)16 << 20));     // 16 MB
  _Float16* fS  = (_Float16*)(ws + ((size_t)32 << 20));     // 64 KB
  _Float16* fC  = fS + 32768;                               // 16 KB
  _Float16* fI  = fC + 8192;                                // 32 KB
  _Float16* fU  = fI + 16384;                               // 16 KB

  prep_x_kernel<<<2048, 256, 0, stream>>>(inputs, xT);
  prep_w_kernel<<<256, 256, 0, stream>>>(w_i2s, w_s2s, w_c2c, w_up, fS, fC, fI, fU);
  lstm_main_kernel<<<32, 256, 0, stream>>>(xT, fS, fC, fI, b_i2s, b_s2s, b_c2c, hid);
  up_proj_kernel<<<8192, 256, 0, stream>>>(hid, fU, b_up, (float*)d_out);
}
